// MambaBlock_36627481100327
// MI455X (gfx1250) — matrix operand order repaired and checked by reading, not compiled, not run on hardware
//
#include <hip/hip_runtime.h>
#include <hip/hip_bf16.h>

// ---------------------------------------------------------------------------
// Mamba block forward for MI455X (gfx1250, wave32, WMMA).
//   - All GEMM operands pre-converted to bf16 (once per buffer, not per use).
//   - GEMM inner loop: branch-free, global_load_b128 + 4x v_wmma per k-step,
//     N-tail handled by pointer clamping (store-side guard only).
//   - Scan / conv / LN: VALU kernels (bandwidth-bound).
// ---------------------------------------------------------------------------

#define BB      4
#define LL      4096
#define DMODEL  1024
#define DINNER  2048
#define DSTATE  16
#define DCONV   4
#define DTRANK  64
#define MROWS   (BB * LL)               // 16384
#define DBCW    (DTRANK + 2 * DSTATE)   // 96

typedef __attribute__((ext_vector_type(16))) __bf16 v16bf;
typedef __attribute__((ext_vector_type(8)))  __bf16 v8bf;   // one 16B load
typedef __attribute__((ext_vector_type(8)))  float  v8f;

union frag16 { v16bf v; struct { v8bf lo, hi; } s; };

__device__ __forceinline__ __bf16 f2bf(float f) {
  // round-to-nearest-even fp32 -> bf16
  unsigned u = __builtin_bit_cast(unsigned, f);
  unsigned r = (u + 0x7FFFu + ((u >> 16) & 1u)) >> 16;
  return __builtin_bit_cast(__bf16, (unsigned short)r);
}

__device__ __forceinline__ float silu_f(float x) {
  return x * __frcp_rn(1.f + __expf(-x));
}

// ---------------------------------------------------------------------------
// Bulk fp32 -> bf16 conversion, 8 elements / thread (n % 2048 == 0).
// ---------------------------------------------------------------------------
__global__ __launch_bounds__(256) void cvt_bf16_kernel(
    const float* __restrict__ in, __bf16* __restrict__ out)
{
  const size_t i = ((size_t)blockIdx.x * blockDim.x + threadIdx.x) * 8;
  const float4 f0 = *(const float4*)(in + i);
  const float4 f1 = *(const float4*)(in + i + 4);
  v8bf o;
  o[0] = f2bf(f0.x); o[1] = f2bf(f0.y); o[2] = f2bf(f0.z); o[3] = f2bf(f0.w);
  o[4] = f2bf(f1.x); o[5] = f2bf(f1.y); o[6] = f2bf(f1.z); o[7] = f2bf(f1.w);
  *(v8bf*)(out + i) = o;
}

// ---------------------------------------------------------------------------
// C = A(MxK bf16, lda) * W^T (W: NxK bf16 row-major); C fp32 MxN (+opt bf16).
// Block = 8 waves (2 wave-rows x 4 wave-cols) -> 32 x 256 block tile.
// Wave: 16 rows x 64 cols = 4 accumulators; per k-step (K += 32):
//   A frag  = 2 x global_load_b128 (bf16 A layout contiguous per lane-half:
//             VGPR0-3 K=half*16+0..7, VGPR4-7 K=half*16+8..15)
//   B frags = 2 x global_load_b128 each (lane n holds K=kk+half*16..+15)
//   4 x v_wmma_f32_16x16x32_bf16 sharing the A fragment, no branches.
// N tail: B pointers clamped to last valid strip; only stores are guarded.
// ---------------------------------------------------------------------------
__global__ __launch_bounds__(256) void gemm_bf16_wmma(
    const __bf16* __restrict__ A, const __bf16* __restrict__ W,
    float* __restrict__ C, __bf16* __restrict__ Cbf,
    int M, int N, int K, int lda,
    const float* __restrict__ bias, int act)
{
  const int lane = threadIdx.x & 31;
  const int wave = threadIdx.x >> 5;
  const int wrow = wave >> 2;               // 0..1
  const int wcol = wave & 3;                // 0..3
  const int row0 = blockIdx.y * 32 + wrow * 16;
  const int col0 = blockIdx.x * 256 + wcol * 64;

  const int m    = lane & 15;               // A row / B col within tile
  const int half = lane >> 4;

  const __bf16* ap = A + (size_t)(row0 + m) * lda + (half << 4);

  // clamp each 16-col strip into [0, N-16] so loads are always in-bounds
  const int cs0c = (col0      < N) ? col0      : (N - 16);
  const int cs1c = (col0 + 16 < N) ? col0 + 16 : (N - 16);
  const int cs2c = (col0 + 32 < N) ? col0 + 32 : (N - 16);
  const int cs3c = (col0 + 48 < N) ? col0 + 48 : (N - 16);
  const __bf16* wp0 = W + (size_t)(cs0c + m) * K + (half << 4);
  const __bf16* wp1 = W + (size_t)(cs1c + m) * K + (half << 4);
  const __bf16* wp2 = W + (size_t)(cs2c + m) * K + (half << 4);
  const __bf16* wp3 = W + (size_t)(cs3c + m) * K + (half << 4);

  v8f c0 = {}, c1 = {}, c2 = {}, c3 = {};

  for (int kk = 0; kk < K; kk += 32) {
    frag16 a, b0, b1, b2, b3;
    // issue all loads of this k-step together (single clause / wait)
    a.s.lo  = *(const v8bf*)(ap + kk);
    a.s.hi  = *(const v8bf*)(ap + kk + 8);
    b0.s.lo = *(const v8bf*)(wp0 + kk);
    b0.s.hi = *(const v8bf*)(wp0 + kk + 8);
    b1.s.lo = *(const v8bf*)(wp1 + kk);
    b1.s.hi = *(const v8bf*)(wp1 + kk + 8);
    b2.s.lo = *(const v8bf*)(wp2 + kk);
    b2.s.hi = *(const v8bf*)(wp2 + kk + 8);
    b3.s.lo = *(const v8bf*)(wp3 + kk);
    b3.s.hi = *(const v8bf*)(wp3 + kk + 8);
    __builtin_prefetch(ap + kk + 32, 0, 1);   // speculative: dropped if OOB

    c0 = __builtin_amdgcn_wmma_f32_16x16x32_bf16(false, a.v, false, b0.v,
                                                 (short)0, c0, false, false);
    c1 = __builtin_amdgcn_wmma_f32_16x16x32_bf16(false, a.v, false, b1.v,
                                                 (short)0, c1, false, false);
    c2 = __builtin_amdgcn_wmma_f32_16x16x32_bf16(false, a.v, false, b2.v,
                                                 (short)0, c2, false, false);
    c3 = __builtin_amdgcn_wmma_f32_16x16x32_bf16(false, a.v, false, b3.v,
                                                 (short)0, c3, false, false);
  }

  // C layout: VGPR v -> row = v + 8*half, col = lane&15.  Store-side guard.
  const v8f acc[4] = {c0, c1, c2, c3};
#pragma unroll
  for (int s = 0; s < 4; ++s) {
    const int cs = col0 + 16 * s;
    if (cs >= N) continue;
    const int cc = cs + m;
#pragma unroll
    for (int v = 0; v < 8; ++v) {
      const int r = row0 + v + (half << 3);
      float val = acc[s][v];
      if (bias) val += bias[cc];
      if (act == 1) val = (val > 20.f) ? val : log1pf(__expf(val));
      C[(size_t)r * N + cc] = val;
      if (Cbf) Cbf[(size_t)r * N + cc] = f2bf(val);
    }
  }
}

// ---------------------------------------------------------------------------
// Depthwise causal conv (width 4) over L + bias + SiLU.
// xz: (B*L, 4096) [xi = first 2048 cols] -> xs fp32 + xs bf16.
// ---------------------------------------------------------------------------
__global__ __launch_bounds__(256) void conv_silu_kernel(
    const float* __restrict__ xz, const float* __restrict__ cw,
    const float* __restrict__ cb, float* __restrict__ xs,
    __bf16* __restrict__ xsb)
{
  const size_t idx = (size_t)blockIdx.x * blockDim.x + threadIdx.x; // M*2048
  const int d  = (int)(idx & (DINNER - 1));
  const size_t bl = idx >> 11;            // b*L + l
  const int l  = (int)(bl & (LL - 1));
  const size_t rowbase = bl - l;          // b*L

  float acc = cb[d];
#pragma unroll
  for (int j = 0; j < DCONV; ++j) {
    const int ll = l - (DCONV - 1) + j;
    if (ll >= 0)
      acc += cw[d * DCONV + j] * xz[(rowbase + ll) * (size_t)(2 * DINNER) + d];
  }
  const float sv = silu_f(acc);
  xs[idx]  = sv;
  xsb[idx] = f2bf(sv);
}

// ---------------------------------------------------------------------------
// Selective scan: one lane per (b, d) channel, 16-state recurrence in VGPRs.
// Fuses y = scan + xs*D then y *= silu(z); writes bf16 for out_proj.
// B/C rows are wave-uniform (broadcast) loads.
// ---------------------------------------------------------------------------
__global__ __launch_bounds__(256) void scan_kernel(
    const float* __restrict__ xs, const float* __restrict__ dtb,
    const float* __restrict__ dbc, const float* __restrict__ xz,
    const float* __restrict__ A_log, const float* __restrict__ Dp,
    __bf16* __restrict__ ybb)
{
  const int idx = blockIdx.x * blockDim.x + threadIdx.x;
  if (idx >= BB * DINNER) return;
  const int b = idx / DINNER;
  const int d = idx & (DINNER - 1);

  float Ad[DSTATE], h[DSTATE];
#pragma unroll
  for (int n = 0; n < DSTATE; ++n) {
    Ad[n] = -__expf(A_log[d * DSTATE + n]);
    h[n]  = 0.f;
  }
  const float Dv = Dp[d];

  for (int l = 0; l < LL; ++l) {
    const size_t row = (size_t)b * LL + l;
    const float dtv = dtb[row * DINNER + d];
    const float xv  = xs[row * DINNER + d];
    const float* bc = dbc + row * DBCW;
    const float dx  = dtv * xv;
    float yacc = 0.f;
#pragma unroll
    for (int n = 0; n < DSTATE; ++n) {
      const float dA = __expf(dtv * Ad[n]);
      h[n] = dA * h[n] + dx * bc[DTRANK + n];
      yacc += h[n] * bc[DTRANK + DSTATE + n];
    }
    yacc += xv * Dv;
    const float zv = xz[row * (size_t)(2 * DINNER) + DINNER + d];
    ybb[row * DINNER + d] = f2bf(yacc * silu_f(zv));
  }
}

// ---------------------------------------------------------------------------
// In-place LayerNorm over 1024 cols; one 256-thread block per row (wave32).
// ---------------------------------------------------------------------------
__global__ __launch_bounds__(256) void ln_kernel(
    float* __restrict__ out, const float* __restrict__ gamma,
    const float* __restrict__ beta)
{
  __shared__ float rs[8], rq[8];
  const int t = threadIdx.x;
  float* p = out + (size_t)blockIdx.x * DMODEL;

  float4 v = *(const float4*)(p + t * 4);
  float s  = v.x + v.y + v.z + v.w;
  float sq = v.x * v.x + v.y * v.y + v.z * v.z + v.w * v.w;
#pragma unroll
  for (int o = 16; o; o >>= 1) {
    s  += __shfl_xor(s, o, 32);
    sq += __shfl_xor(sq, o, 32);
  }
  const int wave = t >> 5, lane = t & 31;
  if (lane == 0) { rs[wave] = s; rq[wave] = sq; }
  __syncthreads();
  if (wave == 0) {
    s  = (lane < 8) ? rs[lane] : 0.f;
    sq = (lane < 8) ? rq[lane] : 0.f;
#pragma unroll
    for (int o = 4; o; o >>= 1) {
      s  += __shfl_xor(s, o, 32);
      sq += __shfl_xor(sq, o, 32);
    }
    if (lane == 0) { rs[0] = s; rq[0] = sq; }
  }
  __syncthreads();
  const float mean = rs[0] * (1.f / DMODEL);
  const float var  = rq[0] * (1.f / DMODEL) - mean * mean;
  const float rstd = rsqrtf(var + 1e-5f);

  const float4 g  = *(const float4*)(gamma + t * 4);
  const float4 bb = *(const float4*)(beta + t * 4);
  v.x = (v.x - mean) * rstd * g.x + bb.x;
  v.y = (v.y - mean) * rstd * g.y + bb.y;
  v.z = (v.z - mean) * rstd * g.z + bb.z;
  v.w = (v.w - mean) * rstd * g.w + bb.w;
  *(float4*)(p + t * 4) = v;
}

// ---------------------------------------------------------------------------
extern "C" void kernel_launch(void* const* d_in, const int* in_sizes, int n_in,
                              void* d_out, int out_size, void* d_ws, size_t ws_size,
                              hipStream_t stream) {
  const float* x          = (const float*)d_in[0];
  const float* in_proj_w  = (const float*)d_in[1];
  const float* conv_w     = (const float*)d_in[2];
  const float* conv_b     = (const float*)d_in[3];
  const float* x_proj_w   = (const float*)d_in[4];
  const float* dt_proj_w  = (const float*)d_in[5];
  const float* dt_proj_b  = (const float*)d_in[6];
  const float* A_log      = (const float*)d_in[7];
  const float* Dp         = (const float*)d_in[8];
  const float* out_proj_w = (const float*)d_in[9];
  const float* ln_g       = (const float*)d_in[10];
  const float* ln_b       = (const float*)d_in[11];
  float* out = (float*)d_out;

  // ---- workspace layout ----
  // fp32 region
  float* xz   = (float*)d_ws;                      // M x 4096
  float* xs   = xz  + (size_t)MROWS * 4096;        // M x 2048
  float* dbc  = xs  + (size_t)MROWS * 2048;        // M x 96
  float* dtb  = dbc + (size_t)MROWS * DBCW;        // M x 2048
  // bf16 region
  __bf16* xbf   = (__bf16*)(dtb + (size_t)MROWS * 2048);  // M x 1024
  __bf16* wbfi  = xbf  + (size_t)MROWS * DMODEL;          // 4096 x 1024
  __bf16* wbfx  = wbfi + (size_t)(2 * DINNER) * DMODEL;   // 96 x 2048
  __bf16* wbfdt = wbfx + (size_t)DBCW * DINNER;           // 2048 x 64
  __bf16* wbfo  = wbfdt + (size_t)DINNER * DTRANK;        // 1024 x 2048
  __bf16* xsb   = wbfo + (size_t)DMODEL * DINNER;         // M x 2048
  __bf16* dbcb  = xsb  + (size_t)MROWS * DINNER;          // M x 96
  __bf16* ybb   = dbcb + (size_t)MROWS * DBCW;            // M x 2048

  const dim3 blk(256);

  // 0) one-time fp32 -> bf16 conversions (8 elems/thread)
  cvt_bf16_kernel<<<(MROWS * DMODEL) / 2048, blk, 0, stream>>>(x, xbf);
  cvt_bf16_kernel<<<(2 * DINNER * DMODEL) / 2048, blk, 0, stream>>>(in_proj_w, wbfi);
  cvt_bf16_kernel<<<(DBCW * DINNER) / 2048, blk, 0, stream>>>(x_proj_w, wbfx);
  cvt_bf16_kernel<<<(DINNER * DTRANK) / 2048, blk, 0, stream>>>(dt_proj_w, wbfdt);
  cvt_bf16_kernel<<<(DMODEL * DINNER) / 2048, blk, 0, stream>>>(out_proj_w, wbfo);

  // 1) xz = x @ in_proj_w^T          (16384 x 4096 x 1024)
  gemm_bf16_wmma<<<dim3(4096 / 256, MROWS / 32), blk, 0, stream>>>(
      xbf, wbfi, xz, nullptr, MROWS, 2 * DINNER, DMODEL, DMODEL, nullptr, 0);

  // 2) xs = silu(causal_conv(xi) + b)  (fp32 + bf16)
  conv_silu_kernel<<<(MROWS * DINNER) / 256, blk, 0, stream>>>(
      xz, conv_w, conv_b, xs, xsb);

  // 3) dbc = xs @ x_proj_w^T         (16384 x 96 x 2048), also bf16 copy
  gemm_bf16_wmma<<<dim3(1, MROWS / 32), blk, 0, stream>>>(
      xsb, wbfx, dbc, dbcb, MROWS, DBCW, DINNER, DINNER, nullptr, 0);

  // 4) dt = softplus(dbc[:, :64] @ dt_proj_w^T + b)   (16384 x 2048 x 64)
  gemm_bf16_wmma<<<dim3(2048 / 256, MROWS / 32), blk, 0, stream>>>(
      dbcb, wbfdt, dtb, nullptr, MROWS, DINNER, DTRANK, DBCW, dt_proj_b, 1);

  // 5) selective scan + D skip + silu(z) gate -> bf16 y
  scan_kernel<<<(BB * DINNER) / 256, blk, 0, stream>>>(
      xs, dtb, dbc, xz, A_log, Dp, ybb);

  // 6) out = y @ out_proj_w^T        (16384 x 1024 x 2048)
  gemm_bf16_wmma<<<dim3(1024 / 256, MROWS / 32), blk, 0, stream>>>(
      ybb, wbfo, out, nullptr, MROWS, DMODEL, DINNER, DINNER, nullptr, 0);

  // 7) LayerNorm in place on d_out
  ln_kernel<<<MROWS, blk, 0, stream>>>(out, ln_g, ln_b);
}
